// NP_82643760709831
// MI455X (gfx1250) — matrix operand order repaired and checked by reading, not compiled, not run on hardware
//
#include <hip/hip_runtime.h>
#include <hip/hip_bf16.h>
#include <stdint.h>

// ---------------------------------------------------------------------------
// Neural Process forward on CDNA5 (gfx1250, wave32).
// Every dense layer runs v_wmma_f32_16x16x32_bf16 with:
//   - all operands pre-converted to bf16 (weights transposed + zero padded)
//   - guard-free inner loop: 4x global_load_b128 + 4x ds_store_b128 staging,
//     2x ds_load_b128 per fragment (layouts per CDNA5 ISA 7.12.2),
//     4 WMMAs per wave per 32-deep K chunk (64x64 tile per 128-thread block).
// ---------------------------------------------------------------------------

#define BB 4
#define NN 1024
#define MM 1024
#define CC 32

typedef __attribute__((ext_vector_type(16))) __bf16 v16bf;
typedef __attribute__((ext_vector_type(8)))  float  v8f;

union Frag { v16bf v; uint4 q[2]; };

__device__ __forceinline__ __bf16 f2bf(float f) {
    union { float f; uint32_t u; } in; in.f = f;
    uint32_t u = in.u;
    uint32_t r = (u + 0x7FFFu + ((u >> 16) & 1u)) >> 16;   // RNE
    union { unsigned short s; __bf16 b; } out; out.s = (unsigned short)r;
    return out.b;
}
__device__ __forceinline__ float bf2f(__bf16 b) {
    union { __bf16 b; unsigned short s; } in; in.b = b;
    union { uint32_t u; float f; } out; out.u = ((uint32_t)in.s) << 16;
    return out.f;
}

// ---------------------------------------------------------------------------
// GEMM: O = act(A(rows x Kp, bf16) @ Wt^T + bias)
//   A      : row-major bf16, stride strideA (= Kp), rows padded to 64
//   Wt     : TRANSPOSED weights, [colsP x Kp] bf16, zero padded
//   omode  : 0 = bf16 out (stride strideO, zero-padded tile)
//            1 = f32 out  (stride cols, guarded store)
//            2 = bf16 out transposed per batch-of-trN rows: [b][colsP][trN]
//   act    : 0 none, 1 relu, 2 (0.1+0.9*sigmoid)*0.1, 3 0.1+0.9*softplus
// Block = 128 threads (4 waves) -> 64x64 output tile, 4 WMMA / wave / chunk.
// ---------------------------------------------------------------------------
__global__ __launch_bounds__(128)
void gemm_bf16(const __bf16* __restrict__ A, long long asb, int strideA,
               const __bf16* __restrict__ Wt, long long wsb, int strideW,
               const float* __restrict__ bias,
               void* __restrict__ Optr, long long osb,
               int rows, int cols, int Kp, int strideO,
               int act, int omode, int trN) {
    __shared__ __bf16 LA[64 * 32];
    __shared__ __bf16 LB[64 * 32];

    const int t = threadIdx.x;
    const int w = t >> 5, lane = t & 31;
    const int g = lane >> 4, n = lane & 15;
    const int srow = t >> 1, skw = (t & 1) * 16;   // staging: 16 bf16 / thread / op

    const __bf16* Ab = A + (size_t)blockIdx.z * asb
                         + (size_t)(blockIdx.y * 64 + srow) * strideA + skw;
    const __bf16* Wb = Wt + (size_t)blockIdx.z * wsb
                          + (size_t)(blockIdx.x * 64 + srow) * strideW + skw;

    v8f acc[4] = {};
    for (int kk = 0; kk < Kp; kk += 32) {
        uint4 a0 = *(const uint4*)(Ab + kk);
        uint4 a1 = *(const uint4*)(Ab + kk + 8);
        uint4 b0 = *(const uint4*)(Wb + kk);
        uint4 b1 = *(const uint4*)(Wb + kk + 8);
        __builtin_prefetch(Ab + kk + 32, 0, 0);    // global_prefetch_b8
        __builtin_prefetch(Wb + kk + 32, 0, 0);
        __syncthreads();
        *(uint4*)(LA + srow * 32 + skw)     = a0;
        *(uint4*)(LA + srow * 32 + skw + 8) = a1;
        *(uint4*)(LB + srow * 32 + skw)     = b0;
        *(uint4*)(LB + srow * 32 + skw + 8) = b1;
        __syncthreads();

        // A fragment (16x32): lane m=n holds K runs [g*8,g*8+8) and [16+g*8,..)
        Frag af;
        af.q[0] = *(const uint4*)(LA + (w * 16 + n) * 32 + g * 8);
        af.q[1] = *(const uint4*)(LA + (w * 16 + n) * 32 + 16 + g * 8);
        // B fragments (32x16): lane (g,n) holds K runs [g*8,g*8+8) and [16+g*8,..) at col n
        #pragma unroll
        for (int f = 0; f < 4; ++f) {
            Frag bf;
            bf.q[0] = *(const uint4*)(LB + (f * 16 + n) * 32 + g * 8);
            bf.q[1] = *(const uint4*)(LB + (f * 16 + n) * 32 + 16 + g * 8);
            acc[f] = __builtin_amdgcn_wmma_f32_16x16x32_bf16(
                         false, af.v, false, bf.v, (short)0, acc[f], false, false);
        }
    }

    // D layout: col = n (lane&15), row = i + 8*g within the wave's 16-row strip
    #pragma unroll
    for (int f = 0; f < 4; ++f) {
        #pragma unroll
        for (int i = 0; i < 8; ++i) {
            int r = blockIdx.y * 64 + w * 16 + i + 8 * g;
            int c = blockIdx.x * 64 + f * 16 + n;
            bool in = (r < rows) && (c < cols);
            float v = 0.0f;
            if (in) {
                v = acc[f][i] + (bias ? bias[c] : 0.0f);
                if (act == 1)      v = fmaxf(v, 0.0f);
                else if (act == 2) v = (0.1f + 0.9f / (1.0f + __expf(-v))) * 0.1f;
                else if (act == 3) {
                    float sp = (v > 20.0f) ? v : log1pf(__expf(v));
                    v = 0.1f + 0.9f * sp;
                }
            }
            if (omode == 0) {
                ((__bf16*)Optr)[(size_t)blockIdx.z * osb + (size_t)r * strideO + c] = f2bf(v);
            } else if (omode == 1) {
                if (in) ((float*)Optr)[(size_t)blockIdx.z * osb + (size_t)r * cols + c] = v;
            } else {  // transposed bf16: [b][colsP][trN]
                int b2 = r / trN, nn = r % trN;
                int colsP = gridDim.x * 64;
                ((__bf16*)Optr)[((size_t)b2 * colsP + c) * trN + nn] = f2bf(v);
            }
        }
    }
}

// ---------------------------------------------------------------------------
// Glue kernels
// ---------------------------------------------------------------------------

// W [K x cols] f32  ->  Wt [colsP x Kp] bf16 (transposed, zero padded)
__global__ void cvt_weight_t(const float* __restrict__ W, __bf16* __restrict__ Wt,
                             int K, int cols, int Kp, int colsP) {
    int idx = blockIdx.x * 256 + threadIdx.x;
    if (idx >= colsP * Kp) return;
    int c = idx / Kp, k = idx % Kp;
    float v = (c < cols && k < K) ? W[(size_t)k * cols + c] : 0.0f;
    Wt[idx] = f2bf(v);
}

__global__ void cvt_f32_bf16(const float* __restrict__ in, __bf16* __restrict__ out, int n) {
    int idx = blockIdx.x * 256 + threadIdx.x;
    if (idx < n) out[idx] = f2bf(in[idx]);
}

// XYb [4096 x 64] bf16: [x_context(32) | y(1) | zeros(31)]
__global__ void concat_xy_bf16(const float* __restrict__ xc, const float* __restrict__ yc,
                               __bf16* __restrict__ XY) {
    int idx = blockIdx.x * 256 + threadIdx.x;        // over 4096*64
    if (idx >= BB * NN * 64) return;
    int row = idx >> 6, c = idx & 63;
    float v = (c < CC) ? xc[row * CC + c] : ((c == CC) ? yc[row] : 0.0f);
    XY[idx] = f2bf(v);
}

// SVb [64 x 256] bf16 = mean over N of SIb [B*N x 256] bf16; rows 4..63 zero
__global__ void mean_rows_bf16(const __bf16* __restrict__ SI, __bf16* __restrict__ S) {
    int idx = blockIdx.x * 256 + threadIdx.x;        // over 64*256
    if (idx >= 64 * 256) return;
    int b = idx >> 8, r = idx & 255;
    float s = 0.0f;
    if (b < BB) {
        for (int nrow = 0; nrow < NN; ++nrow)
            s += bf2f(SI[((size_t)b * NN + nrow) * 256 + r]);
        s *= (1.0f / NN);
    }
    S[idx] = f2bf(s);
}

__global__ void make_z(const float* __restrict__ mu, const float* __restrict__ var,
                       const float* __restrict__ eps, float* __restrict__ z) {
    int i = blockIdx.x * 256 + threadIdx.x;
    if (i < BB * 128) z[i] = mu[i] + var[i] * eps[i];
}

// scores[b,m,n] = -sum_d |k[b,n,d] - q[m,d]|  (bf16 inputs, stride 64)
__global__ void laplace_scores(const __bf16* __restrict__ Kc, const __bf16* __restrict__ Q,
                               float* __restrict__ SC) {
    __shared__ float q[CC];
    int b = blockIdx.z, mrow = blockIdx.y, nidx = blockIdx.x * 256 + threadIdx.x;
    if (threadIdx.x < CC) q[threadIdx.x] = bf2f(Q[mrow * 64 + threadIdx.x]);
    __syncthreads();
    const __bf16* krow = Kc + ((size_t)b * NN + nidx) * 64;
    float s = 0.0f;
    #pragma unroll
    for (int d = 0; d < CC; ++d) s += fabsf(bf2f(krow[d]) - q[d]);
    SC[((size_t)b * MM + mrow) * NN + nidx] = -s;
}

// softmax over rows of 1024; f32 scratch in, bf16 weights out
__global__ void softmax_bf16(float* __restrict__ SC, __bf16* __restrict__ SCb) {
    __shared__ float red[256];
    float*   p = SC  + (size_t)blockIdx.x * NN;
    __bf16*  o = SCb + (size_t)blockIdx.x * NN;
    int t = threadIdx.x;
    float mx = -3.4e38f;
    for (int i = t; i < NN; i += 256) mx = fmaxf(mx, p[i]);
    red[t] = mx; __syncthreads();
    for (int s = 128; s > 0; s >>= 1) { if (t < s) red[t] = fmaxf(red[t], red[t + s]); __syncthreads(); }
    mx = red[0]; __syncthreads();
    float sum = 0.0f;
    for (int i = t; i < NN; i += 256) { float e = __expf(p[i] - mx); p[i] = e; sum += e; }
    red[t] = sum; __syncthreads();
    for (int s = 128; s > 0; s >>= 1) { if (t < s) red[t] += red[t + s]; __syncthreads(); }
    float inv = 1.0f / red[0];
    for (int i = t; i < NN; i += 256) o[i] = f2bf(p[i] * inv);
}

// ZXb[b,m,:] = [r_sample(128) | z(128) | x_grid[m](32)] bf16; z_sample f32 out
__global__ void build_zx(const float* __restrict__ rs, const float* __restrict__ z,
                         const float* __restrict__ xg, __bf16* __restrict__ ZX,
                         float* __restrict__ zs_out) {
    int bm = blockIdx.x;
    int b = bm / MM, mrow = bm % MM;
    int t = threadIdx.x;                              // 128 threads
    float zv = z[b * 128 + t];
    ZX[(size_t)bm * 288 + t]       = f2bf(rs[(size_t)bm * 128 + t]);
    ZX[(size_t)bm * 288 + 128 + t] = f2bf(zv);
    zs_out[(size_t)bm * 128 + t]   = zv;
    if (t < CC) ZX[(size_t)bm * 288 + 256 + t] = f2bf(xg[mrow * CC + t]);
}

// ---------------------------------------------------------------------------
// Host orchestration
// ---------------------------------------------------------------------------
static inline int r32(int x) { return (x + 31) & ~31; }
static inline int r64(int x) { return (x + 63) & ~63; }

static inline void gemm(hipStream_t s,
                        const __bf16* A, long long asb, int strideA,
                        const __bf16* Wt, long long wsb, const float* bias,
                        void* O, long long osb,
                        int rows, int K, int cols, int strideO,
                        int act, int omode, int batch, int trN = 1) {
    int Kp = r32(K), colsP = r64(cols), rowsP = r64(rows);
    dim3 grid(colsP / 64, rowsP / 64, batch);
    gemm_bf16<<<grid, 128, 0, s>>>(A, asb, strideA, Wt, wsb, Kp, bias,
                                   O, osb, rows, cols, Kp, strideO, act, omode, trN);
}

// layer table: h4 h5 h6 s_dense s_mu s_logvar h1 h2 h3 h7 h8 h9 h10 s g1 g2 g3 g4 gm gv
static const int LK[20]    = { 33,512,512,256,256,256, 33,512,512, 32,512, 32,512,256,288,512,512,512,512,512 };
static const int LCOLS[20] = {512,512,256,256,128,128,512,512,256,512, 32,512, 32,128,512,512,512,512,  1,  1 };

extern "C" void kernel_launch(void* const* d_in, const int* in_sizes, int n_in,
                              void* d_out, int out_size, void* d_ws, size_t ws_size,
                              hipStream_t stream) {
    const float* xc  = (const float*)d_in[0];
    const float* yc  = (const float*)d_in[1];
    const float* xg  = (const float*)d_in[2];
    const float* eps = (const float*)d_in[3];
    auto Wp = [&](int i) { return (const float*)d_in[4 + 2 * i]; };
    auto Bp = [&](int i) { return (const float*)d_in[5 + 2 * i]; };

    // f32 scratch
    float* SC = (float*)d_ws;                     // 4*1024*1024
    float* ZB = SC + (size_t)4 * 1024 * 1024;     // 512
    // bf16 scratch (16B aligned: 16MB + 2KB from base)
    __bf16* bp   = (__bf16*)(ZB + 512);
    __bf16* XYb  = bp;                 bp += (size_t)4096 * 64;
    __bf16* AB0  = bp;                 bp += (size_t)4096 * 512;
    __bf16* AB1  = bp;                 bp += (size_t)4096 * 512;
    __bf16* SIb  = bp;                 bp += (size_t)4096 * 256;
    __bf16* RIt  = bp;                 bp += (size_t)4 * 256 * 1024;   // [b][256][1024]
    __bf16* KBb  = bp;                 bp += (size_t)4096 * 64;
    __bf16* QBb  = bp;                 bp += (size_t)1024 * 64;
    __bf16* XCb  = bp;                 bp += (size_t)4096 * 32;
    __bf16* XGb  = bp;                 bp += (size_t)1024 * 32;
    __bf16* SCb  = bp;                 bp += (size_t)4 * 1024 * 1024;
    __bf16* RBb  = bp;                 bp += (size_t)4096 * 256;
    __bf16* SVb  = bp;                 bp += (size_t)64 * 256;
    __bf16* SV2b = bp;                 bp += (size_t)64 * 256;
    __bf16* ZXb  = bp;                 bp += (size_t)4096 * 288;
    __bf16* wt[20];
    for (int i = 0; i < 20; ++i) { wt[i] = bp; bp += (size_t)r64(LCOLS[i]) * r32(LK[i]); }

    float* out     = (float*)d_out;
    float* o_ymean = out;                 // [4096]
    float* o_sigma = out + 4096;          // [4096]
    float* o_mu    = out + 8192;          // [512]
    float* o_var   = out + 8704;          // [512]
    float* o_rs    = out + 9216;          // [4096*128]
    float* o_zs    = out + 533504;        // [4096*128]

    // ---- weight convert (transpose + pad + bf16)
    for (int i = 0; i < 20; ++i) {
        int Kp = r32(LK[i]), cP = r64(LCOLS[i]);
        int ntot = cP * Kp;
        cvt_weight_t<<<(ntot + 255) / 256, 256, 0, stream>>>(Wp(i), wt[i], LK[i], LCOLS[i], Kp, cP);
    }
    cvt_f32_bf16<<<(4096 * 32 + 255) / 256, 256, 0, stream>>>(xc, XCb, 4096 * 32);
    cvt_f32_bf16<<<(1024 * 32 + 255) / 256, 256, 0, stream>>>(xg, XGb, 1024 * 32);
    concat_xy_bf16<<<(4096 * 64 + 255) / 256, 256, 0, stream>>>(xc, yc, XYb);

    // ---- latent path
    gemm(stream, XYb, 0, 64,  wt[0], 0, Bp(0), AB0, 0, 4096,  33, 512, 512, 1, 0, 1);
    gemm(stream, AB0, 0, 512, wt[1], 0, Bp(1), AB1, 0, 4096, 512, 512, 512, 1, 0, 1);
    gemm(stream, AB1, 0, 512, wt[2], 0, Bp(2), SIb, 0, 4096, 512, 256, 256, 1, 0, 1);
    mean_rows_bf16<<<(64 * 256 + 255) / 256, 256, 0, stream>>>(SIb, SVb);
    gemm(stream, SVb,  0, 256, wt[3], 0, Bp(3), SV2b, 0, 4, 256, 256, 256, 1, 0, 1);
    gemm(stream, SV2b, 0, 256, wt[4], 0, Bp(4), o_mu,  0, 4, 256, 128, 128, 0, 1, 1);
    gemm(stream, SV2b, 0, 256, wt[5], 0, Bp(5), o_var, 0, 4, 256, 128, 128, 2, 1, 1);
    make_z<<<2, 256, 0, stream>>>(o_mu, o_var, eps, ZB);

    // ---- determinate path (h3 stores r_i transposed per batch: [b][256][1024])
    gemm(stream, XYb, 0, 64,  wt[6], 0, Bp(6), AB0, 0, 4096,  33, 512, 512, 1, 0, 1);
    gemm(stream, AB0, 0, 512, wt[7], 0, Bp(7), AB1, 0, 4096, 512, 512, 512, 1, 0, 1);
    gemm(stream, AB1, 0, 512, wt[8], 0, Bp(8), RIt, 0, 4096, 512, 256, 256, 1, 2, 1, 1024);
    gemm(stream, XCb, 0, 32,  wt[9],  0, Bp(9),  AB0, 0, 4096,  32, 512, 512, 1, 0, 1);
    gemm(stream, AB0, 0, 512, wt[10], 0, Bp(10), KBb, 0, 4096, 512,  32,  64, 1, 0, 1);
    gemm(stream, XGb, 0, 32,  wt[11], 0, Bp(11), AB1, 0, 1024,  32, 512, 512, 1, 0, 1);
    gemm(stream, AB1, 0, 512, wt[12], 0, Bp(12), QBb, 0, 1024, 512,  32,  64, 1, 0, 1);

    // ---- Laplace attention
    laplace_scores<<<dim3(NN / 256, MM, BB), 256, 0, stream>>>(KBb, QBb, SC);
    softmax_bf16<<<BB * MM, 256, 0, stream>>>(SC, SCb);
    gemm(stream, SCb, (long long)MM * NN, 1024, RIt, (long long)256 * 1024, nullptr,
         RBb, (long long)MM * 256, MM, NN, 256, 256, 0, 0, BB);
    gemm(stream, RBb, 0, 256, wt[13], 0, Bp(13), o_rs, 0, 4096, 256, 128, 128, 0, 1, 1);

    // ---- decoder
    build_zx<<<BB * MM, 128, 0, stream>>>(o_rs, ZB, xg, ZXb, o_zs);
    gemm(stream, ZXb, 0, 288, wt[14], 0, Bp(14), AB0, 0, 4096, 288, 512, 512, 1, 0, 1);
    gemm(stream, AB0, 0, 512, wt[15], 0, Bp(15), AB1, 0, 4096, 512, 512, 512, 1, 0, 1);
    gemm(stream, AB1, 0, 512, wt[16], 0, Bp(16), AB0, 0, 4096, 512, 512, 512, 1, 0, 1);
    gemm(stream, AB0, 0, 512, wt[17], 0, Bp(17), AB1, 0, 4096, 512, 512, 512, 1, 0, 1);
    gemm(stream, AB1, 0, 512, wt[18], 0, Bp(18), o_ymean, 0, 4096, 512, 1, 1, 0, 1, 1);
    gemm(stream, AB1, 0, 512, wt[19], 0, Bp(19), o_sigma, 0, 4096, 512, 1, 1, 3, 1, 1);
}
